// TopoPool_49881750176015
// MI455X (gfx1250) — compile-verified, matrix operand order checked by reading, not run on hardware
//
#include <hip/hip_runtime.h>
#include <hip/hip_bf16.h>
#include <math.h>

// ---------------------------------------------------------------------------
// TopoPool for gfx1250 (MI455X).  Bandwidth/graph-bound workload; the GEMV
// z = x @ W^T uses v_wmma_f32_16x16x32_f16 (16 rows per wave, W in column 0
// of B, staged in LDS as f16).  BFS watershed fill runs a fixed 64
// level-synchronous rounds so the launch sequence is graph-capture safe.
// ---------------------------------------------------------------------------

typedef __attribute__((ext_vector_type(16))) _Float16 v16h;
typedef __attribute__((ext_vector_type(8)))  float    v8f;

#define BMAX 1024          // max batch segments supported
#define LEVELS 64          // fixed BFS depth (graph capture: no readback)
#define NEG_HUGE (-3.402823466e38f)

// ---- helpers --------------------------------------------------------------

__device__ __forceinline__ void atomicMaxFloat(float* addr, float val) {
  if (!(val == val)) return;  // skip NaN
  if (val >= 0.0f) {
    atomicMax((int*)addr, __float_as_int(val));
  } else {
    atomicMin((unsigned int*)addr, __float_as_uint(val));
  }
}

__global__ void fill_f32(float* p, float v, int n) {
  int i = blockIdx.x * blockDim.x + threadIdx.x;
  if (i < n) p[i] = v;
}
__global__ void fill_i32(int* p, int v, int n) {
  int i = blockIdx.x * blockDim.x + threadIdx.x;
  if (i < n) p[i] = v;
}

// ---- 1) GEMV z = x @ W^T + b via WMMA -------------------------------------
// One wave = 16 rows of x. A(16x32 f16) from x via float4 loads, B(32x16 f16)
// with W (LDS-staged f16) packed into column 0 (lanes 0 and 16 per the B
// lane-striping).  D column 0 holds the 16 dot products in lane 0 (M=0..7)
// and lane 16 (M=8..15).  Compile-time C -> fully unrolled 8x WMMA chain.

template <int CT>
__global__ void gemv_wmma_fixed(const float* __restrict__ x,
                                const float* __restrict__ W,
                                const float* __restrict__ bias,
                                float* __restrict__ z, int N) {
  __shared__ _Float16 wlds[CT];
  // stage W once per block (before any wave exits: barrier must see all waves)
  for (int k = threadIdx.x; k < CT; k += blockDim.x) wlds[k] = (_Float16)W[k];
  __syncthreads();

  int wave = threadIdx.x >> 5;
  int lane = threadIdx.x & 31;
  int half = lane >> 4;       // which K-half of the lane-striped layout
  int mrow = lane & 15;
  int base = blockIdx.x * 128 + wave * 16;
  if (base >= N) return;      // wave-uniform exit; EXEC all-ones for WMMA
  int row = base + mrow;
  if (row >= N) row = N - 1;  // clamp tail loads (stores are guarded)

  const float* xrow = x + (size_t)row * CT;
  v8f acc = {};
  constexpr int KSTEPS = CT / 32;
#pragma unroll
  for (int ks = 0; ks < KSTEPS; ++ks) {
    const int k32 = ks * 32;
    __builtin_prefetch(xrow + k32 + 64, 0, 1);   // global_prefetch_b8 ahead
    // A fragment: 16 f16 per lane from two 32B vector loads
    const float4* xq = (const float4*)(xrow + k32) + half * 2;
    float4 q0 = xq[0], q1 = xq[1];      // K = k32 + half*8 + 0..7
    float4 q2 = xq[4], q3 = xq[5];      // K = k32 + 16 + half*8 + 0..7
    v16h a;
    a[0]  = (_Float16)q0.x; a[1]  = (_Float16)q0.y;
    a[2]  = (_Float16)q0.z; a[3]  = (_Float16)q0.w;
    a[4]  = (_Float16)q1.x; a[5]  = (_Float16)q1.y;
    a[6]  = (_Float16)q1.z; a[7]  = (_Float16)q1.w;
    a[8]  = (_Float16)q2.x; a[9]  = (_Float16)q2.y;
    a[10] = (_Float16)q2.z; a[11] = (_Float16)q2.w;
    a[12] = (_Float16)q3.x; a[13] = (_Float16)q3.y;
    a[14] = (_Float16)q3.z; a[15] = (_Float16)q3.w;
    // B fragment: column N=0 only, lives in lanes 0 and 16
    v16h bm;
#pragma unroll
    for (int j = 0; j < 16; ++j) bm[j] = (_Float16)0.0f;
    if (mrow == 0) {
      const _Float16* wr = wlds + k32 + half * 8;
#pragma unroll
      for (int j = 0; j < 8; ++j) { bm[j] = wr[j]; bm[8 + j] = wr[16 + j]; }
    }
    acc = __builtin_amdgcn_wmma_f32_16x16x32_f16(
        false, a, false, bm, (short)0, acc, false, false);
  }
  float b0 = bias[0];
  if (lane == 0) {
#pragma unroll
    for (int r = 0; r < 8; ++r) { int rr = base + r;     if (rr < N) z[rr] = acc[r] + b0; }
  } else if (lane == 16) {
#pragma unroll
    for (int r = 0; r < 8; ++r) { int rr = base + 8 + r; if (rr < N) z[rr] = acc[r] + b0; }
  }
}

// generic fallback for C != 256 (runtime trip count, W from global)
__global__ void gemv_wmma(const float* __restrict__ x, const float* __restrict__ W,
                          const float* __restrict__ bias, float* __restrict__ z,
                          int N, int C) {
  int wave = threadIdx.x >> 5;
  int lane = threadIdx.x & 31;
  int half = lane >> 4;
  int mrow = lane & 15;
  int base = blockIdx.x * 128 + wave * 16;
  if (base >= N) return;
  int row = base + mrow;
  if (row >= N) row = N - 1;

  v8f acc = {};
  int ksteps = C >> 5;
  const float* xrow = x + (size_t)row * C;
  for (int ks = 0; ks < ksteps; ++ks) {
    int k32 = ks * 32;
    const float* xr = xrow + k32 + half * 8;
    v16h a, bm;
#pragma unroll
    for (int j = 0; j < 8; ++j) {
      a[j]     = (_Float16)xr[j];
      a[8 + j] = (_Float16)xr[16 + j];
    }
#pragma unroll
    for (int j = 0; j < 16; ++j) bm[j] = (_Float16)0.0f;
    if (mrow == 0) {
      const float* wr = W + k32 + half * 8;
#pragma unroll
      for (int j = 0; j < 8; ++j) {
        bm[j]     = (_Float16)wr[j];
        bm[8 + j] = (_Float16)wr[16 + j];
      }
    }
    acc = __builtin_amdgcn_wmma_f32_16x16x32_f16(
        false, a, false, bm, (short)0, acc, false, false);
  }
  float b0 = bias[0];
  if (lane == 0) {
#pragma unroll
    for (int r = 0; r < 8; ++r) { int rr = base + r;     if (rr < N) z[rr] = acc[r] + b0; }
  } else if (lane == 16) {
#pragma unroll
    for (int r = 0; r < 8; ++r) { int rr = base + 8 + r; if (rr < N) z[rr] = acc[r] + b0; }
  }
}

// ---- 2) batch segment softmax -> elev -------------------------------------

__global__ void batch_maxk(const float* __restrict__ z, const int* __restrict__ batch,
                           float* __restrict__ m, int N) {
  int i = blockIdx.x * blockDim.x + threadIdx.x;
  if (i < N) atomicMaxFloat(&m[batch[i]], z[i]);
}
__global__ void expsumk(const float* __restrict__ z, const int* __restrict__ batch,
                        const float* __restrict__ m, float* __restrict__ s,
                        float* __restrict__ elev, int N) {
  int i = blockIdx.x * blockDim.x + threadIdx.x;
  if (i < N) {
    float e = expf(z[i] - m[batch[i]]);
    elev[i] = e;
    atomicAdd(&s[batch[i]], e);
  }
}
__global__ void normk(float* __restrict__ elev, const int* __restrict__ batch,
                      const float* __restrict__ s, int N) {
  int i = blockIdx.x * blockDim.x + threadIdx.x;
  if (i < N) elev[i] /= s[batch[i]];
}

// ---- 3) edge stats: in-degree, >=-count, descending out-degree ------------

__global__ void edge_stats(const int* __restrict__ ei, int E,
                           const float* __restrict__ elev,
                           int* __restrict__ deg, int* __restrict__ ge,
                           int* __restrict__ outdeg) {
  int j = blockIdx.x * blockDim.x + threadIdx.x;
  if (j >= E) return;
  int sN = ei[j], dN = ei[E + j];
  atomicAdd(&deg[dN], 1);
  float es = elev[sN], ed = elev[dN];
  if (ed >= es) atomicAdd(&ge[dN], 1);
  if (ed <= es) atomicAdd(&outdeg[sN], 1);   // descending edge s->d
}
__global__ void peak_flagk(const int* __restrict__ deg, const int* __restrict__ ge,
                           int* __restrict__ flag, int N) {
  int i = blockIdx.x * blockDim.x + threadIdx.x;
  if (i < N) flag[i] = (deg[i] == ge[i]) ? 1 : 0;
}

// ---- exclusive scan (block scan + serial block-sum scan + add-back) -------

__global__ void scan_block(const int* __restrict__ in, int* __restrict__ out,
                           int* __restrict__ bsums, int n) {
  __shared__ int sh[256];
  int t = threadIdx.x;
  int g = blockIdx.x * 256 + t;
  int v = (g < n) ? in[g] : 0;
  sh[t] = v;
  __syncthreads();
  for (int off = 1; off < 256; off <<= 1) {
    int y = (t >= off) ? sh[t - off] : 0;
    __syncthreads();
    sh[t] += y;
    __syncthreads();
  }
  if (g < n) out[g] = sh[t] - v;       // exclusive
  if (t == 255) bsums[blockIdx.x] = sh[255];
}
__global__ void scan_tail(int* __restrict__ bsums, int nb, int* __restrict__ total) {
  int run = 0;
  for (int i = 0; i < nb; ++i) { int t = bsums[i]; bsums[i] = run; run += t; }
  *total = run;
}
__global__ void scan_addoff(int* __restrict__ out, const int* __restrict__ bsums, int n) {
  int g = blockIdx.x * 256 + threadIdx.x;
  if (g < n) out[g] += bsums[blockIdx.x];
}

// ---- 4) peak collection: seed clusters, frontier 0 ------------------------

__global__ void collect_peaks(const int* __restrict__ flag, const int* __restrict__ pid,
                              const float* __restrict__ elev, const float* __restrict__ x,
                              const int* __restrict__ batch, int N, int C,
                              int* __restrict__ visited, float* __restrict__ max_elev,
                              float* __restrict__ pooled, float* __restrict__ cbf,
                              int* __restrict__ cbi, int2* __restrict__ fr0, int P) {
  int i = blockIdx.x * blockDim.x + threadIdx.x;
  if (i >= N || !flag[i]) return;
  int c = pid[i];
  if (c >= P) return;
  visited[i]  = 1;
  max_elev[c] = elev[i];
  cbf[c] = (float)batch[i];
  cbi[c] = batch[i];
  fr0[c] = make_int2(i, c);
  const float* xr = x + (size_t)i * C;
  float* pr = pooled + (size_t)c * C;
  for (int k = 0; k < C; ++k) pr[k] = xr[k];
}
__global__ void set_counts(int* cnts, int P) { cnts[0] = P; cnts[1] = 0; }

// ---- CSR fill for descending edges ----------------------------------------

__global__ void csr_fill(const int* __restrict__ ei, int E, const float* __restrict__ elev,
                         const int* __restrict__ rowoff, int* __restrict__ cursor,
                         int* __restrict__ colidx) {
  int j = blockIdx.x * blockDim.x + threadIdx.x;
  if (j >= E) return;
  int sN = ei[j], dN = ei[E + j];
  if (elev[dN] <= elev[sN]) {
    int pos = rowoff[sN] + atomicAdd(&cursor[sN], 1);
    colidx[pos] = dN;
  }
}

// ---- 5) BFS level kernels --------------------------------------------------

__global__ void bfs_expand(const int2* __restrict__ fin, const int* __restrict__ cntIn,
                           int2* __restrict__ fout, int* __restrict__ cntOut,
                           const int* __restrict__ rowoff, const int* __restrict__ colidx,
                           const int* __restrict__ visited, int* __restrict__ claim,
                           int level, int maxf) {
  int n = *cntIn; if (n > maxf) n = maxf;
  int stride = gridDim.x * blockDim.x;
  for (int i = blockIdx.x * blockDim.x + threadIdx.x; i < n; i += stride) {
    int fn = fin[i].x, fc = fin[i].y;
    int e0 = rowoff[fn], e1 = rowoff[fn + 1];
    int key = ((level + 1) << 24) | fc;   // fc < 2^17 << 2^24
    for (int e = e0; e < e1; ++e) {
      int ed = colidx[e];
      if (visited[ed]) continue;          // visited == levels < current only
      if (claim[ed] == key) continue;     // fast duplicate reject
      int old = atomicExch(&claim[ed], key);
      if (old == key) continue;           // lost the (node,cluster) race
      int p = atomicAdd(cntOut, 1);
      if (p < maxf) fout[p] = make_int2(ed, fc);
    }
  }
}
__global__ void bfs_finalize(const int2* __restrict__ fr, const int* __restrict__ cnt,
                             int maxf, int* __restrict__ visited,
                             float* __restrict__ max_elev, const float* __restrict__ elev,
                             float* __restrict__ pooled, const float* __restrict__ x, int C) {
  int n = *cnt; if (n > maxf) n = maxf;
  for (int p = blockIdx.x; p < n; p += gridDim.x) {
    int node = fr[p].x, clus = fr[p].y;
    if (threadIdx.x == 0) {
      visited[node] = 1;
      atomicMaxFloat(&max_elev[clus], elev[node]);
    }
    const float* xr = x + (size_t)node * C;
    float* pr = pooled + (size_t)clus * C;
    for (int c = threadIdx.x; c < C; c += blockDim.x) atomicMaxFloat(&pr[c], xr[c]);
  }
}
__global__ void bfs_reset(int* cnts, int curSlot, int nxtSlot, int maxf) {
  if (cnts[nxtSlot] > maxf) cnts[nxtSlot] = maxf;
  cnts[curSlot] = 0;
}

// ---- 6) cluster softmax + final scaling -----------------------------------

__global__ void cmaxk(const float* __restrict__ max_elev, const int* __restrict__ cbi,
                      float* __restrict__ m2, int P) {
  int p = blockIdx.x * blockDim.x + threadIdx.x;
  if (p < P) atomicMaxFloat(&m2[cbi[p]], max_elev[p]);
}
__global__ void cexpsumk(const float* __restrict__ max_elev, const int* __restrict__ cbi,
                         const float* __restrict__ m2, float* __restrict__ s2,
                         float* __restrict__ eC, int P) {
  int p = blockIdx.x * blockDim.x + threadIdx.x;
  if (p < P) {
    float e = expf(max_elev[p] - m2[cbi[p]]);
    eC[p] = e;
    atomicAdd(&s2[cbi[p]], e);
  }
}
__global__ void pooled_scale(float* __restrict__ pooled, const float* __restrict__ eC,
                             const float* __restrict__ s2, const int* __restrict__ cbi,
                             int P, int C) {
  int idx = blockIdx.x * blockDim.x + threadIdx.x;
  if (idx < P * C) {
    int p = idx / C;
    pooled[idx] *= eC[p] / s2[cbi[p]];
  }
}

// ---------------------------------------------------------------------------

extern "C" void kernel_launch(void* const* d_in, const int* in_sizes, int n_in,
                              void* d_out, int out_size, void* d_ws, size_t ws_size,
                              hipStream_t stream) {
  const float* x     = (const float*)d_in[0];
  const int*   ei    = (const int*)d_in[1];
  const int*   batch = (const int*)d_in[2];
  const float* W     = (const float*)d_in[3];
  const float* bias  = (const float*)d_in[4];

  const int C = in_sizes[3];            // W is [1, C]
  const int N = in_sizes[0] / C;
  const int E = in_sizes[1] / 2;
  int P = (out_size - N) / (C + 1);     // pooled P*C + cb P + elev N
  if (P < 1) P = 1;

  float* pooled = (float*)d_out;
  float* cbf    = pooled + (size_t)P * C;
  float* elev   = cbf + P;

  // ---- carve workspace ----
  char* wp = (char*)d_ws;
  size_t remain = ws_size;
  auto carve = [&](size_t bytes) -> void* {
    size_t rb = (bytes + 255) & ~(size_t)255;
    void* p = (void*)wp;
    wp += rb;
    remain = (remain > rb) ? (remain - rb) : 0;
    return p;
  };
  float* z        = (float*)carve((size_t)N * 4);
  float* m        = (float*)carve(BMAX * 4);
  float* s        = (float*)carve(BMAX * 4);
  float* m2       = (float*)carve(BMAX * 4);
  float* s2       = (float*)carve(BMAX * 4);
  int*   deg      = (int*)carve((size_t)N * 4);
  int*   ge       = (int*)carve((size_t)N * 4);
  int*   flag     = (int*)carve((size_t)N * 4);
  int*   pid      = (int*)carve((size_t)N * 4);
  int*   bsums    = (int*)carve(4096 * 4);
  int*   totalP   = (int*)carve(256);
  int*   outdeg   = (int*)carve((size_t)N * 4);
  int*   rowoff   = (int*)carve((size_t)(N + 1) * 4);
  int*   cursor   = (int*)carve((size_t)N * 4);
  int*   visited  = (int*)carve((size_t)N * 4);
  int*   claim    = (int*)carve((size_t)N * 4);
  int*   cbi      = (int*)carve((size_t)P * 4);
  float* max_elev = (float*)carve((size_t)P * 4);
  float* eC       = (float*)carve((size_t)P * 4);
  int*   frCnt    = (int*)carve(256);
  int*   colidx   = (int*)carve((size_t)E * 4);
  // remaining budget -> ping-pong frontiers of (node,cluster) pairs
  long long maxfL = (long long)(remain / (2 * sizeof(int2)));
  int MAXF = (int)(maxfL > (4 << 20) ? (4 << 20) : maxfL);
  if (MAXF < 1024) MAXF = 1024;
  int2* frA = (int2*)carve((size_t)MAXF * sizeof(int2));
  int2* frB = (int2*)carve((size_t)MAXF * sizeof(int2));

  const int T = 256;
  int gN = (N + T - 1) / T;
  int gE = (E + T - 1) / T; if (gE < 1) gE = 1;
  int gP = (P + T - 1) / T;
  int gPC = ((P * C) + T - 1) / T;
  int NB = (N + 255) / 256;

  // ---- init ----
  fill_f32<<<(BMAX + T - 1) / T, T, 0, stream>>>(m,  NEG_HUGE, BMAX);
  fill_f32<<<(BMAX + T - 1) / T, T, 0, stream>>>(s,  0.0f,     BMAX);
  fill_f32<<<(BMAX + T - 1) / T, T, 0, stream>>>(m2, NEG_HUGE, BMAX);
  fill_f32<<<(BMAX + T - 1) / T, T, 0, stream>>>(s2, 0.0f,     BMAX);
  fill_i32<<<gN, T, 0, stream>>>(deg,     0, N);
  fill_i32<<<gN, T, 0, stream>>>(ge,      0, N);
  fill_i32<<<gN, T, 0, stream>>>(outdeg,  0, N);
  fill_i32<<<gN, T, 0, stream>>>(cursor,  0, N);
  fill_i32<<<gN, T, 0, stream>>>(visited, 0, N);
  fill_i32<<<gN, T, 0, stream>>>(claim,  -1, N);

  // ---- 1) GEMV (WMMA) ----
  if (C == 256) {
    gemv_wmma_fixed<256><<<(N + 127) / 128, 256, 0, stream>>>(x, W, bias, z, N);
  } else {
    gemv_wmma<<<(N + 127) / 128, 256, 0, stream>>>(x, W, bias, z, N, C);
  }

  // ---- 2) batch softmax -> elev ----
  batch_maxk<<<gN, T, 0, stream>>>(z, batch, m, N);
  expsumk<<<gN, T, 0, stream>>>(z, batch, m, s, elev, N);
  normk<<<gN, T, 0, stream>>>(elev, batch, s, N);

  // ---- 3) peaks + descending CSR ----
  edge_stats<<<gE, T, 0, stream>>>(ei, E, elev, deg, ge, outdeg);
  peak_flagk<<<gN, T, 0, stream>>>(deg, ge, flag, N);

  scan_block<<<NB, 256, 0, stream>>>(flag, pid, bsums, N);
  scan_tail<<<1, 1, 0, stream>>>(bsums, NB, totalP);
  scan_addoff<<<NB, 256, 0, stream>>>(pid, bsums, N);

  collect_peaks<<<gN, T, 0, stream>>>(flag, pid, elev, x, batch, N, C,
                                      visited, max_elev, pooled, cbf, cbi, frA, P);
  set_counts<<<1, 1, 0, stream>>>(frCnt, P);

  scan_block<<<NB, 256, 0, stream>>>(outdeg, rowoff, bsums, N);
  scan_tail<<<1, 1, 0, stream>>>(bsums, NB, &rowoff[N]);
  scan_addoff<<<NB, 256, 0, stream>>>(rowoff, bsums, N);
  csr_fill<<<gE, T, 0, stream>>>(ei, E, elev, rowoff, cursor, colidx);

  // ---- 5) fixed-depth level-synchronous BFS ----
  for (int l = 0; l < LEVELS; ++l) {
    int cur = l & 1, nxt = (l + 1) & 1;
    const int2* fin = cur ? frB : frA;
    int2* fout      = nxt ? frB : frA;
    bfs_expand<<<1024, 256, 0, stream>>>(fin, frCnt + cur, fout, frCnt + nxt,
                                         rowoff, colidx, visited, claim, l, MAXF);
    bfs_finalize<<<2048, 256, 0, stream>>>(fout, frCnt + nxt, MAXF, visited,
                                           max_elev, elev, pooled, x, C);
    bfs_reset<<<1, 1, 0, stream>>>(frCnt, cur, nxt, MAXF);
  }

  // ---- 6) cluster softmax + scale pooled ----
  cmaxk<<<gP, T, 0, stream>>>(max_elev, cbi, m2, P);
  cexpsumk<<<gP, T, 0, stream>>>(max_elev, cbi, m2, s2, eC, P);
  pooled_scale<<<gPC, T, 0, stream>>>(pooled, eC, s2, cbi, P, C);
}
